// ProteinFlow_73254962201141
// MI455X (gfx1250) — compile-verified, matrix operand order checked by reading, not run on hardware
//
#include <hip/hip_runtime.h>
#include <hip/hip_bf16.h>
#include <math.h>

#define B_ 2
#define N_ 384
#define CS 256
#define CZ 128
#define H_ 8
#define CHID 128
#define PQ_ 8
#define PV_ 12
#define NBLK 6
#define TFL 2
#define TFH 4
#define FEAT 64
#define NBINS 22
#define BN (B_*N_)
#define BNN ((long)B_*N_*N_)
#define INFV 100000.0f

typedef __attribute__((ext_vector_type(16))) _Float16 v16h;
typedef __attribute__((ext_vector_type(8)))  float    v8f;

__device__ __forceinline__ float wsum(float x){
#pragma unroll
  for (int o=16;o>0;o>>=1) x += __shfl_xor(x,o,32);
  return x;
}
__device__ __forceinline__ float wmaxr(float x){
#pragma unroll
  for (int o=16;o>0;o>>=1) x = fmaxf(x,__shfl_xor(x,o,32));
  return x;
}
__device__ __forceinline__ int imin(int a,int b){ return a<b?a:b; }

// ---------------------------------------------------------------------------
// Fragment loaders.  ISA 16-bit A layout (wave32):
//   lanes 0-15:  e<8 -> K=k0+e,    e>=8 -> K=k0+16+(e-8)
//   lanes 16-31: e<8 -> K=k0+8+e,  e>=8 -> K=k0+24+(e-8)
// -> two contiguous 8-float runs per lane (k0+half*8, k0+16+half*8) =
//    four global_load_b128 per fragment.
// ---------------------------------------------------------------------------
__device__ __forceinline__ v16h load_row_frag(const float* __restrict__ rowp,
                                              int k0, int half, float alpha){
  const float4* p0 = (const float4*)(rowp + k0 + half*8);
  const float4* p1 = (const float4*)(rowp + k0 + 16 + half*8);
  float4 x0 = p0[0], x1 = p0[1], y0 = p1[0], y1 = p1[1];
  v16h f;
  f[0]=(_Float16)(x0.x*alpha); f[1]=(_Float16)(x0.y*alpha);
  f[2]=(_Float16)(x0.z*alpha); f[3]=(_Float16)(x0.w*alpha);
  f[4]=(_Float16)(x1.x*alpha); f[5]=(_Float16)(x1.y*alpha);
  f[6]=(_Float16)(x1.z*alpha); f[7]=(_Float16)(x1.w*alpha);
  f[8]=(_Float16)(y0.x*alpha); f[9]=(_Float16)(y0.y*alpha);
  f[10]=(_Float16)(y0.z*alpha); f[11]=(_Float16)(y0.w*alpha);
  f[12]=(_Float16)(y1.x*alpha); f[13]=(_Float16)(y1.y*alpha);
  f[14]=(_Float16)(y1.z*alpha); f[15]=(_Float16)(y1.w*alpha);
  return f;
}
// branch-free tail: clamped address + select
__device__ __forceinline__ v16h load_row_tail(const float* __restrict__ rowp,
                                              int k0, int half, int K, float alpha){
  v16h f;
#pragma unroll
  for (int e=0;e<16;++e){
    int k = k0 + e + half*8 + ((e>=8)?8:0);
    float v = rowp[imin(k, K-1)];
    f[e] = (_Float16)((k < K) ? v*alpha : 0.0f);
  }
  return f;
}
__device__ __forceinline__ v16h load_col_frag(const float* __restrict__ B, int ldb,
                                              int col, int k0, int half, int K){
  v16h f;
#pragma unroll
  for (int e=0;e<16;++e){
    int k = k0 + e + half*8 + ((e>=8)?8:0);
    float v = B[(long)imin(k,K-1)*ldb + col];
    f[e] = (_Float16)((k < K) ? v : 0.0f);
  }
  return f;
}

// ---------------------------------------------------------------------------
// Fast WMMA GEMM: 64x32 output per wave (4 A-frags x 2 B-frags = 8 WMMAs
// per 32-wide K-chunk).  Requires M%64==0, Nn%32==0, lda%4==0
// (and ldb%4==0 when TB=1).  SC=0 -> alpha statically 1.0 (folds the
// per-element v_pk_mul away); SC=1 -> runtime alpha applied to A.
// ---------------------------------------------------------------------------
template<int TB, int SC>
__global__ void __launch_bounds__(32)
wmma_gemm_fast(const float* __restrict__ A, long sAo, long sAi, int lda,
               const float* __restrict__ Bm, long sBo, long sBi, int ldb,
               const float* __restrict__ bias, float alpha, int act,
               float* __restrict__ C, long sCo, long sCi, int ldc,
               int K, int inner)
{
  int bz = blockIdx.z;
  int oi = bz / inner, ii = bz % inner;
  A  += (long)oi*sAo + (long)ii*sAi;
  Bm += (long)oi*sBo + (long)ii*sBi;
  C  += (long)oi*sCo + (long)ii*sCi;

  const float as = SC ? alpha : 1.0f;
  int lane = threadIdx.x;
  int half = lane >> 4, l15 = lane & 15;
  int m0 = blockIdx.x * 64, n0 = blockIdx.y * 32;
  const float* Ar[4];
#pragma unroll
  for (int t=0;t<4;++t) Ar[t] = A + (long)(m0 + t*16 + l15)*lda;
  const float* Bc0 = TB ? (Bm + (long)(n0 + l15)*ldb)      : Bm;
  const float* Bc1 = TB ? (Bm + (long)(n0 + 16 + l15)*ldb) : Bm;

  v8f acc[8];   // acc[at*2+bt]
#pragma unroll
  for (int t=0;t<8;++t){
    float bv = bias ? bias[n0 + (t&1)*16 + l15] : 0.0f;
#pragma unroll
    for (int v=0;v<8;++v) acc[t][v] = bv;
  }

  int kfull = K & ~31;
  for (int k0 = 0; k0 < kfull; k0 += 32) {
    v16h a0 = load_row_frag(Ar[0], k0, half, as);
    v16h a1 = load_row_frag(Ar[1], k0, half, as);
    v16h a2 = load_row_frag(Ar[2], k0, half, as);
    v16h a3 = load_row_frag(Ar[3], k0, half, as);
    v16h b0, b1;
    if (TB) { b0 = load_row_frag(Bc0, k0, half, 1.0f);
              b1 = load_row_frag(Bc1, k0, half, 1.0f); }
    else    { b0 = load_col_frag(Bm, ldb, n0 + l15,      k0, half, K);
              b1 = load_col_frag(Bm, ldb, n0 + 16 + l15, k0, half, K); }
    acc[0] = __builtin_amdgcn_wmma_f32_16x16x32_f16(false,a0,false,b0,(short)0,acc[0],false,false);
    acc[1] = __builtin_amdgcn_wmma_f32_16x16x32_f16(false,a0,false,b1,(short)0,acc[1],false,false);
    acc[2] = __builtin_amdgcn_wmma_f32_16x16x32_f16(false,a1,false,b0,(short)0,acc[2],false,false);
    acc[3] = __builtin_amdgcn_wmma_f32_16x16x32_f16(false,a1,false,b1,(short)0,acc[3],false,false);
    acc[4] = __builtin_amdgcn_wmma_f32_16x16x32_f16(false,a2,false,b0,(short)0,acc[4],false,false);
    acc[5] = __builtin_amdgcn_wmma_f32_16x16x32_f16(false,a2,false,b1,(short)0,acc[5],false,false);
    acc[6] = __builtin_amdgcn_wmma_f32_16x16x32_f16(false,a3,false,b0,(short)0,acc[6],false,false);
    acc[7] = __builtin_amdgcn_wmma_f32_16x16x32_f16(false,a3,false,b1,(short)0,acc[7],false,false);
  }
  if (kfull < K) {                    // branch-free K tail
    int k0 = kfull;
    v16h a0 = load_row_tail(Ar[0], k0, half, K, as);
    v16h a1 = load_row_tail(Ar[1], k0, half, K, as);
    v16h a2 = load_row_tail(Ar[2], k0, half, K, as);
    v16h a3 = load_row_tail(Ar[3], k0, half, K, as);
    v16h b0, b1;
    if (TB) { b0 = load_row_tail(Bc0, k0, half, K, 1.0f);
              b1 = load_row_tail(Bc1, k0, half, K, 1.0f); }
    else    { b0 = load_col_frag(Bm, ldb, n0 + l15,      k0, half, K);
              b1 = load_col_frag(Bm, ldb, n0 + 16 + l15, k0, half, K); }
    acc[0] = __builtin_amdgcn_wmma_f32_16x16x32_f16(false,a0,false,b0,(short)0,acc[0],false,false);
    acc[1] = __builtin_amdgcn_wmma_f32_16x16x32_f16(false,a0,false,b1,(short)0,acc[1],false,false);
    acc[2] = __builtin_amdgcn_wmma_f32_16x16x32_f16(false,a1,false,b0,(short)0,acc[2],false,false);
    acc[3] = __builtin_amdgcn_wmma_f32_16x16x32_f16(false,a1,false,b1,(short)0,acc[3],false,false);
    acc[4] = __builtin_amdgcn_wmma_f32_16x16x32_f16(false,a2,false,b0,(short)0,acc[4],false,false);
    acc[5] = __builtin_amdgcn_wmma_f32_16x16x32_f16(false,a2,false,b1,(short)0,acc[5],false,false);
    acc[6] = __builtin_amdgcn_wmma_f32_16x16x32_f16(false,a3,false,b0,(short)0,acc[6],false,false);
    acc[7] = __builtin_amdgcn_wmma_f32_16x16x32_f16(false,a3,false,b1,(short)0,acc[7],false,false);
  }
#pragma unroll
  for (int at=0; at<4; ++at)
#pragma unroll
    for (int v=0; v<8; ++v){
      int cm = m0 + at*16 + half*8 + v;
#pragma unroll
      for (int bt=0; bt<2; ++bt){
        float r = acc[at*2+bt][v];
        if (act == 1) r = fmaxf(r, 0.0f);
        C[(long)cm*ldc + n0 + bt*16 + l15] = r;
      }
    }
}

// ---------------------------------------------------------------------------
// Guarded generic GEMM (small / odd shapes), templated on TB, fully
// branch-free loads via clamp+select.
// ---------------------------------------------------------------------------
template<int TB>
__global__ void __launch_bounds__(32)
wmma_gemm_g(const float* __restrict__ A, long sAo, long sAi, int lda,
            const float* __restrict__ Bm, long sBo, long sBi, int ldb,
            const float* __restrict__ bias, float alpha, int act,
            float* __restrict__ C, long sCo, long sCi, int ldc,
            int M, int Nn, int K, int inner)
{
  int bz = blockIdx.z;
  int oi = bz / inner, ii = bz % inner;
  A  += (long)oi*sAo + (long)ii*sAi;
  Bm += (long)oi*sBo + (long)ii*sBi;
  C  += (long)oi*sCo + (long)ii*sCi;

  int lane = threadIdx.x;
  int half = lane >> 4, l15 = lane & 15;
  int m0 = blockIdx.x * 16, n0 = blockIdx.y * 16;
  int arow = m0 + l15, bcol = n0 + l15;
  bool aok = arow < M, bok = bcol < Nn;
  int arc = imin(arow, M-1), bcc = imin(bcol, Nn-1);
  const float* Arow = A + (long)arc*lda;
  const float* Brow = TB ? (Bm + (long)bcc*ldb) : Bm;

  v8f acc;
  {
    float bv = bias ? bias[bcc] : 0.0f;
    bv = (bias && bok) ? bv : 0.0f;
#pragma unroll
    for (int v=0;v<8;++v) acc[v] = bv;
  }
  for (int k0 = 0; k0 < K; k0 += 32) {
    v16h af, bf;
#pragma unroll
    for (int e = 0; e < 16; ++e) {
      int k = k0 + e + half*8 + ((e >= 8) ? 8 : 0);
      int kc = imin(k, K-1);
      float av = Arow[kc];
      af[e] = (_Float16)((aok && k < K) ? av*alpha : 0.0f);
      float bv = TB ? Brow[kc] : Bm[(long)kc*ldb + bcc];
      bf[e] = (_Float16)((bok && k < K) ? bv : 0.0f);
    }
    acc = __builtin_amdgcn_wmma_f32_16x16x32_f16(
        false, af, false, bf, (short)0, acc, false, false);
  }
#pragma unroll
  for (int v = 0; v < 8; ++v) {
    int cm = m0 + half*8 + v;
    if (cm < M && bok) {
      float r = acc[v];
      if (act == 1) r = fmaxf(r, 0.0f);
      C[(long)cm*ldc + bcol] = r;
    }
  }
}

// ---------------------------------------------------------------------------
// Elementwise / reduction kernels (wave32)
// ---------------------------------------------------------------------------
__global__ void __launch_bounds__(256)
init_pose_k(const float* trans, const float* rot, float* tr, float* R){
  int i = blockIdx.x*256 + threadIdx.x;
  if (i < BN*3) tr[i] = trans[i]*0.1f;
  if (i < BN*9) R[i]  = rot[i];
}

__global__ void __launch_bounds__(256)
node_input_k(const float* t, const float* mask, float* out){
  int idx = blockIdx.x*256 + threadIdx.x;
  if (idx >= BN*256) return;
  int c = idx & 255; int n = (idx >> 8) % N_; int b = idx / (N_*256);
  float m = mask[b*N_+n], v;
  if (c < 128) {
    int K = (c < 64) ? c : c-64;
    float ang = (float)n * 3.14159265358979f / powf(2056.0f, 2.0f*K/128.0f);
    v = (c < 64) ? sinf(ang) : cosf(ang);
  } else {
    int cc = c - 128; int K = (cc < 64) ? cc : cc-64;
    float ts  = t[b] * 2056.0f;
    float emb = expf((float)K * (-logf(2056.0f)/63.0f));
    float ang = ts * emb;
    v = (cc < 64) ? sinf(ang) : cosf(ang);
  }
  out[idx] = v * m;
}

__global__ void __launch_bounds__(256)
relpos_build_k(float* out){
  long idx = (long)blockIdx.x*256 + threadIdx.x;
  if (idx >= BNN*FEAT) return;
  int c = idx % FEAT; long r = idx / FEAT;
  int j = r % N_; int i = (r / N_) % N_;
  float rel = (float)(i - j);
  int K = (c < 32) ? c : c-32;
  float ang = rel * 3.14159265358979f / powf(2056.0f, 2.0f*K/64.0f);
  out[idx] = (c < 32) ? sinf(ang) : cosf(ang);
}

__global__ void __launch_bounds__(256)
edge_feat_k(const float* p_i, const float* rp, const float* pos, float* out){
  long idx = (long)blockIdx.x*256 + threadIdx.x;
  if (idx >= BNN) return;
  int j = idx % N_; int i = (idx / N_) % N_; int b = idx / ((long)N_*N_);
  float* o = out + idx*236;
  const float* pi_ = p_i + ((long)b*N_+i)*64;
  const float* pj_ = p_i + ((long)b*N_+j)*64;
  const float* rpp = rp + idx*64;
  for (int c = 0; c < 64; ++c){ o[c]=pi_[c]; o[64+c]=pj_[c]; o[128+c]=rpp[c]; }
  const float* a = pos + ((long)b*N_+i)*3;
  const float* bb = pos + ((long)b*N_+j)*3;
  float dx=a[0]-bb[0], dy=a[1]-bb[1], dz=a[2]-bb[2];
  float d = sqrtf(dx*dx+dy*dy+dz*dz);
  const float step = (20.0f-0.001f)/21.0f;
  for (int k = 0; k < NBINS; ++k){
    float lo = 0.001f + k*step;
    float hi = (k < 21) ? (0.001f + (k+1)*step) : 1e8f;
    o[192+k] = (d > lo && d < hi) ? 1.0f : 0.0f;
    o[214+k] = 0.0f;
  }
}

__global__ void __launch_bounds__(32)
ln_k(const float* x, const float* res, const float* g, const float* be,
     const float* rmask, int maskmode, float* out, int D){
  long row = blockIdx.x; int lane = threadIdx.x;
  const float* xr = x + row*D;
  const float* rr = res ? res + row*D : (const float*)0;
  float s=0.f, s2=0.f;
  for (int c = lane; c < D; c += 32){
    float v = xr[c] + (rr ? rr[c] : 0.0f); s += v; s2 += v*v;
  }
  s = wsum(s); s2 = wsum(s2);
  float mean = s / D, var = s2 / D - mean*mean;
  float inv = rsqrtf(var + 1e-5f);
  float m = 1.0f;
  if (maskmode == 1) m = rmask[row];
  else if (maskmode == 2){
    int b = row / ((long)N_*N_); int i = (row / N_) % N_; int j = row % N_;
    m = rmask[b*N_+i]*rmask[b*N_+j];
  }
  for (int c = lane; c < D; c += 32){
    float v = xr[c] + (rr ? rr[c] : 0.0f);
    out[row*D + c] = ((v - mean)*inv*g[c] + be[c]) * m;
  }
}

__global__ void __launch_bounds__(32)
softmax_k(float* a, int L){
  long row = blockIdx.x; int lane = threadIdx.x;
  float* p = a + row*(long)L;
  float mx = -3.0e38f;
  for (int c = lane; c < L; c += 32) mx = fmaxf(mx, p[c]);
  mx = wmaxr(mx);
  float s = 0.f;
  for (int c = lane; c < L; c += 32){ float e = __expf(p[c]-mx); p[c]=e; s+=e; }
  s = wsum(s);
  float inv = 1.0f / s;
  for (int c = lane; c < L; c += 32) p[c] *= inv;
}

__global__ void __launch_bounds__(256)
to_pts_k(const float* lin, const float* R, const float* tr, float* out, int npts){
  int idx = blockIdx.x*256 + threadIdx.x;
  if (idx >= BN*npts) return;
  int p = idx % npts; int r = idx / npts;
  const float* L = lin + (long)r*3*npts;
  float x0=L[0*npts+p], x1=L[1*npts+p], x2=L[2*npts+p];
  const float* Rr = R + r*9; const float* t = tr + r*3;
  float* o = out + (long)idx*3;
  o[0] = Rr[0]*x0 + Rr[1]*x1 + Rr[2]*x2 + t[0];
  o[1] = Rr[3]*x0 + Rr[4]*x1 + Rr[5]*x2 + t[1];
  o[2] = Rr[6]*x0 + Rr[7]*x1 + Rr[8]*x2 + t[2];
}

__global__ void __launch_bounds__(256)
inv_rot_k(const float* opt, const float* R, const float* tr, float* out){
  int idx = blockIdx.x*256 + threadIdx.x;
  if (idx >= BN*H_*PV_) return;
  int r = idx / (H_*PV_);
  const float* Rr = R + r*9; const float* t = tr + r*3;
  const float* p = opt + (long)idx*3; float* o = out + (long)idx*3;
  float d0=p[0]-t[0], d1=p[1]-t[1], d2=p[2]-t[2];
  o[0] = Rr[0]*d0 + Rr[3]*d1 + Rr[6]*d2;
  o[1] = Rr[1]*d0 + Rr[4]*d1 + Rr[7]*d2;
  o[2] = Rr[2]*d0 + Rr[5]*d1 + Rr[8]*d2;
}

__global__ void __launch_bounds__(256)
ipa_score_fix_k(float* att, const float* bz, const float* qpts,
                const float* kvpts, const float* head_w, const float* mask){
  long idx = (long)blockIdx.x*256 + threadIdx.x;
  if (idx >= (long)B_*H_*N_*N_) return;
  int j = idx % N_;
  int i = (idx / N_) % N_;
  int h = (idx / ((long)N_*N_)) % H_;
  int b = idx / ((long)H_*N_*N_);
  float s = att[idx];
  s += 0.5773502691896258f * bz[(((long)(b*N_+i)*N_)+j)*H_ + h];
  const float* qp = qpts  + ((long)(b*N_+i)*H_ + h)*PQ_*3;
  const float* kp = kvpts + ((long)(b*N_+j)*H_ + h)*(PQ_+PV_)*3;
  float d2 = 0.f;
#pragma unroll
  for (int p = 0; p < PQ_; ++p){
    float dx=qp[p*3+0]-kp[p*3+0], dy=qp[p*3+1]-kp[p*3+1], dz=qp[p*3+2]-kp[p*3+2];
    d2 += dx*dx + dy*dy + dz*dz;
  }
  float hw = log1pf(__expf(head_w[h])) * 0.09622504486493763f;
  s += -0.5f * hw * d2;
  s += INFV * (mask[b*N_+i]*mask[b*N_+j] - 1.0f);
  att[idx] = s;
}

__global__ void __launch_bounds__(256)
key_pad_k(float* att, const float* mask){
  long idx = (long)blockIdx.x*256 + threadIdx.x;
  if (idx >= (long)B_*TFH*N_*N_) return;
  int j = idx % N_;
  int b = idx / ((long)TFH*N_*N_);
  if (mask[b*N_+j] == 0.0f) att[idx] = -1.0e9f;
}

__global__ void __launch_bounds__(256)
ipa_cat_k(const float* o, const float* opt2, const float* opair, float* cat){
  long idx = (long)blockIdx.x*256 + threadIdx.x;
  if (idx >= (long)BN*2432) return;
  int c = idx % 2432; long r = idx / 2432;
  float v;
  if (c < 1024) v = o[r*1024 + c];
  else if (c < 1312){ int q=c-1024; int x=q/96; int hp=q%96; v = opt2[(r*96+hp)*3 + x]; }
  else if (c < 1408){ int hp=c-1312; const float* p = opt2 + (r*96+hp)*3;
                      v = sqrtf(p[0]*p[0]+p[1]*p[1]+p[2]*p[2] + 1e-8f); }
  else v = opair[r*1024 + (c-1408)];
  cat[idx] = v;
}

__global__ void __launch_bounds__(256)
ecat_k(const float* z, const float* ne, float* out){
  long idx = (long)blockIdx.x*256 + threadIdx.x;
  if (idx >= BNN*384) return;
  int c = idx % 384; long r = idx / 384;
  int j = r % N_; long bi = r / N_; int i = bi % N_; int b = bi / N_;
  float v;
  if (c < 128)      v = z[r*128 + c];
  else if (c < 256) v = ne[((long)b*N_+i)*128 + (c-128)];
  else              v = ne[((long)b*N_+j)*128 + (c-256)];
  out[idx] = v;
}

__global__ void __launch_bounds__(256)
add_k(float* out, const float* a, const float* b, long n){
  long i = (long)blockIdx.x*256 + threadIdx.x;
  if (i < n) out[i] = a[i] + b[i];
}

__global__ void __launch_bounds__(256)
scale_rows_k(float* x, const float* m, long rows, int D){
  long i = (long)blockIdx.x*256 + threadIdx.x;
  if (i < rows*(long)D) x[i] *= m[i / D];
}

__global__ void __launch_bounds__(256)
bb_update_k(const float* upd, float* tr, float* R){
  int idx = blockIdx.x*256 + threadIdx.x;
  if (idx >= BN) return;
  const float* u = upd + idx*6;
  float qx=u[0], qy=u[1], qz=u[2], tx=u[3], ty=u[4], tz=u[5];
  float inv = rsqrtf(1.0f + qx*qx + qy*qy + qz*qz);
  float w = inv, x = qx*inv, y = qy*inv, z = qz*inv;
  float Q[9] = { 1-2*(y*y+z*z), 2*(x*y-w*z), 2*(x*z+w*y),
                 2*(x*y+w*z), 1-2*(x*x+z*z), 2*(y*z-w*x),
                 2*(x*z-w*y), 2*(y*z+w*x), 1-2*(x*x+y*y) };
  float* Rr = R + idx*9; float* t = tr + idx*3;
  float R0[9];
#pragma unroll
  for (int i2=0;i2<9;++i2) R0[i2]=Rr[i2];
  t[0] += R0[0]*tx + R0[1]*ty + R0[2]*tz;
  t[1] += R0[3]*tx + R0[4]*ty + R0[5]*tz;
  t[2] += R0[6]*tx + R0[7]*ty + R0[8]*tz;
#pragma unroll
  for (int i2=0;i2<3;++i2)
#pragma unroll
    for (int k=0;k<3;++k)
      Rr[i2*3+k] = R0[i2*3+0]*Q[k] + R0[i2*3+1]*Q[3+k] + R0[i2*3+2]*Q[6+k];
}

__global__ void __launch_bounds__(256)
write_out_k(const float* tr, const float* R, float* out){
  int i = blockIdx.x*256 + threadIdx.x;
  if (i < BN*3) out[i] = tr[i]*10.0f;
  if (i < BN*9) out[BN*3 + i] = R[i];
}

// ---------------------------------------------------------------------------
// Host-side orchestration
// ---------------------------------------------------------------------------
struct Lin { const float *b, *w; };
struct LN2 { const float *b, *g; };
struct TfLayer { Lin ff1, ff2, in_proj; LN2 ln1, ln2; Lin out_proj; };
struct Block {
  Lin bb;
  Lin et_final, et_init; LN2 et_ln; Lin et_t1, et_t2; bool has_et;
  Lin ipa_b; const float* head_w; Lin kv, kv_pts, out, q, q_pts;
  LN2 ipa_ln; Lin post; TfLayer tf[TFL];
  Lin tl1, tl2, tl3; LN2 tln;
};

static inline bool fast_ok(long M, int Nn, int lda, int ldb, int transB){
  return (M % 64 == 0) && (Nn % 32 == 0) && (lda % 4 == 0) &&
         (!transB || (ldb % 4 == 0));
}

static void gemm_any(hipStream_t st,
                     const float* A, long sAo, long sAi, int lda,
                     const float* Bm, long sBo, long sBi, int ldb, int transB,
                     const float* bias, float alpha, int act,
                     float* C, long sCo, long sCi, int ldc,
                     long M, int Nn, int K, int outer, int inner){
  unsigned bz = (unsigned)(outer*inner);
  bool sc = (alpha != 1.0f);
  if (fast_ok(M, Nn, lda, ldb, transB)) {
    dim3 g((unsigned)(M/64), (unsigned)(Nn/32), bz);
    if (transB) {
      if (sc) wmma_gemm_fast<1,1><<<g,32,0,st>>>(A,sAo,sAi,lda, Bm,sBo,sBi,ldb,
                                                 bias,alpha,act, C,sCo,sCi,ldc, K, inner);
      else    wmma_gemm_fast<1,0><<<g,32,0,st>>>(A,sAo,sAi,lda, Bm,sBo,sBi,ldb,
                                                 bias,alpha,act, C,sCo,sCi,ldc, K, inner);
    } else {
      if (sc) wmma_gemm_fast<0,1><<<g,32,0,st>>>(A,sAo,sAi,lda, Bm,sBo,sBi,ldb,
                                                 bias,alpha,act, C,sCo,sCi,ldc, K, inner);
      else    wmma_gemm_fast<0,0><<<g,32,0,st>>>(A,sAo,sAi,lda, Bm,sBo,sBi,ldb,
                                                 bias,alpha,act, C,sCo,sCi,ldc, K, inner);
    }
  } else {
    dim3 g((unsigned)((M+15)/16), (unsigned)((Nn+15)/16), bz);
    if (transB)
      wmma_gemm_g<1><<<g,32,0,st>>>(A,sAo,sAi,lda, Bm,sBo,sBi,ldb,
                                    bias,alpha,act, C,sCo,sCi,ldc,
                                    (int)M,Nn,K, inner);
    else
      wmma_gemm_g<0><<<g,32,0,st>>>(A,sAo,sAi,lda, Bm,sBo,sBi,ldb,
                                    bias,alpha,act, C,sCo,sCi,ldc,
                                    (int)M,Nn,K, inner);
  }
}

static inline void lin_launch(hipStream_t st, const float* X, const Lin& L,
                              float* Y, long M, int K, int Nn, int act){
  gemm_any(st, X,0,0,K, L.w,0,0,Nn,0, L.b,1.0f,act, Y,0,0,Nn, M,Nn,K, 1,1);
}
static inline void bgemm(hipStream_t st, const float* A, long sAo, long sAi, int lda,
                         const float* Bm, long sBo, long sBi, int ldb, int transB,
                         float alpha, float* C, long sCo, long sCi, int ldc,
                         int M, int Nn, int K, int outer, int inner){
  gemm_any(st, A,sAo,sAi,lda, Bm,sBo,sBi,ldb,transB, nullptr,alpha,0,
           C,sCo,sCi,ldc, M,Nn,K, outer,inner);
}
static inline dim3 eg(long n){ return dim3((unsigned)((n+255)/256)); }

extern "C" void kernel_launch(void* const* d_in, const int* in_sizes, int n_in,
                              void* d_out, int out_size, void* d_ws, size_t ws_size,
                              hipStream_t stream) {
  (void)in_sizes; (void)n_in; (void)out_size; (void)ws_size;
  const float* t_in  = (const float*)d_in[0];
  const float* mask  = (const float*)d_in[1];
  const float* trans = (const float*)d_in[2];
  const float* rot   = (const float*)d_in[3];

  // ---- decode params in JAX pytree (sorted-key) leaf order -----------------
  int pi = 4;
  auto nf = [&]() -> const float* { return (const float*)d_in[pi++]; };
  auto rdlin = [&]() -> Lin { Lin L; L.b = nf(); L.w = nf(); return L; };
  auto rdln  = [&]() -> LN2 { LN2 L; L.b = nf(); L.g = nf(); return L; };

  Block blk[NBLK];
  for (int b = 0; b < NBLK; ++b) {
    Block& K = blk[b];
    K.bb = rdlin();
    K.has_et = (b < NBLK-1);
    if (K.has_et) { // edge_tr keys sorted: final, init, ln, t1, t2
      K.et_final = rdlin(); K.et_init = rdlin(); K.et_ln = rdln();
      K.et_t1 = rdlin(); K.et_t2 = rdlin();
    }
    // ipa keys sorted: b, head_w, kv, kv_pts, out, q, q_pts
    K.ipa_b = rdlin(); K.head_w = nf();
    K.kv = rdlin(); K.kv_pts = rdlin(); K.out = rdlin();
    K.q = rdlin(); K.q_pts = rdlin();
    K.ipa_ln = rdln();
    K.post = rdlin();
    for (int l = 0; l < TFL; ++l) { // keys: ff1, ff2, in_proj, ln1, ln2, out_proj
      TfLayer& T = K.tf[l];
      T.ff1 = rdlin(); T.ff2 = rdlin(); T.in_proj = rdlin();
      T.ln1 = rdln(); T.ln2 = rdln(); T.out_proj = rdlin();
    }
    K.tl1 = rdlin(); K.tl2 = rdlin(); K.tl3 = rdlin(); K.tln = rdln();
  }
  // edge keys sorted: l1, l2, l3, ln, relpos, s_p
  Lin e_l1 = rdlin(), e_l2 = rdlin(), e_l3 = rdlin();
  LN2 e_ln = rdln();
  Lin e_rel = rdlin(), e_sp = rdlin();
  Lin node_lin = rdlin();

  // ---- workspace bump allocator -------------------------------------------
  char* wsp = (char*)d_ws; size_t off = 0;
  auto alloc = [&](size_t nflt) -> float* {
    float* p = (float*)(wsp + off);
    off += nflt * sizeof(float);
    off = (off + 255) & ~(size_t)255;
    return p;
  };
  float* tr    = alloc(BN*3);
  float* R     = alloc(BN*9);
  float* nin   = alloc((size_t)BN*256);
  float* s     = alloc((size_t)BN*256);
  float* p_i   = alloc((size_t)BN*64);
  float* rp_in = alloc((size_t)BNN*64);
  float* rp_out= alloc((size_t)BNN*64);
  float* feat  = alloc((size_t)BNN*236);
  float* eh1   = alloc((size_t)BNN*128);
  float* eh2   = alloc((size_t)BNN*128);
  float* z     = alloc((size_t)BNN*128);
  float* qb    = alloc((size_t)BN*1024);
  float* kvb   = alloc((size_t)BN*2048);
  float* qpl   = alloc((size_t)BN*192);
  float* kvpl  = alloc((size_t)BN*480);
  float* qpts  = alloc((size_t)BN*192);
  float* kvpts = alloc((size_t)BN*480);
  float* bzb   = alloc((size_t)BNN*8);
  float* att   = alloc((size_t)B_*H_*N_*N_);
  float* ob    = alloc((size_t)BN*1024);
  float* opt   = alloc((size_t)BN*288);
  float* opt2  = alloc((size_t)BN*288);
  float* opair = alloc((size_t)BN*1024);
  float* cat   = alloc((size_t)BN*2432);
  float* t1b   = alloc((size_t)BN*256);
  float* t2b   = alloc((size_t)BN*256);
  float* qkv   = alloc((size_t)BN*768);
  float* hbuf  = alloc((size_t)BN*256);
  float* att4  = alloc((size_t)B_*TFH*N_*N_);
  float* obuf4 = alloc((size_t)BN*256);
  float* updb  = alloc((size_t)BN*6);
  float* neb   = alloc((size_t)BN*128);
  float* ecat  = alloc((size_t)BNN*384);
  float* gh1   = alloc((size_t)BNN*384);
  float* gh2   = alloc((size_t)BNN*384);

  // ---- forward -------------------------------------------------------------
  init_pose_k<<<eg(BN*9),256,0,stream>>>(trans, rot, tr, R);
  node_input_k<<<eg((long)BN*256),256,0,stream>>>(t_in, mask, nin);
  lin_launch(stream, nin, node_lin, s, BN, 256, 256, 0);

  // edge embedder
  lin_launch(stream, s, e_sp, p_i, BN, 256, 64, 0);
  relpos_build_k<<<eg(BNN*64),256,0,stream>>>(rp_in);
  lin_launch(stream, rp_in, e_rel, rp_out, BNN, 64, 64, 0);
  edge_feat_k<<<eg(BNN),256,0,stream>>>(p_i, rp_out, trans, feat);
  lin_launch(stream, feat, e_l1, eh1, BNN, 236, 128, 1);
  lin_launch(stream, eh1, e_l2, eh2, BNN, 128, 128, 1);
  lin_launch(stream, eh2, e_l3, eh1, BNN, 128, 128, 0);
  ln_k<<<dim3((unsigned)BNN),32,0,stream>>>(eh1, nullptr, e_ln.g, e_ln.b,
                                            mask, 2, z, 128);
  scale_rows_k<<<eg((long)BN*256),256,0,stream>>>(s, mask, BN, 256);

  const float qk_alpha = 0.05103103630798288f;  // sqrt(1/(3*128))
  for (int b = 0; b < NBLK; ++b) {
    Block& K = blk[b];
    // -------- IPA --------
    lin_launch(stream, s, K.q,      qb,   BN, 256, 1024, 0);
    lin_launch(stream, s, K.kv,     kvb,  BN, 256, 2048, 0);
    lin_launch(stream, s, K.q_pts,  qpl,  BN, 256, 192,  0);
    to_pts_k<<<eg(BN*64),256,0,stream>>>(qpl, R, tr, qpts, 64);
    lin_launch(stream, s, K.kv_pts, kvpl, BN, 256, 480,  0);
    to_pts_k<<<eg(BN*160),256,0,stream>>>(kvpl, R, tr, kvpts, 160);
    lin_launch(stream, z, K.ipa_b,  bzb,  BNN, 128, 8,   0);
    bgemm(stream, qb,  (long)N_*1024, 128, 1024,
                  kvb, (long)N_*2048, 256, 2048, 1, qk_alpha,
                  att, (long)H_*N_*N_, (long)N_*N_, N_,
                  N_, N_, 128, B_, H_);
    ipa_score_fix_k<<<eg((long)B_*H_*N_*N_),256,0,stream>>>(att, bzb, qpts, kvpts,
                                                            K.head_w, mask);
    softmax_k<<<dim3((unsigned)(B_*H_*N_)),32,0,stream>>>(att, N_);
    bgemm(stream, att, (long)H_*N_*N_, (long)N_*N_, N_,
                  kvb+128, (long)N_*2048, 256, 2048, 0, 1.0f,
                  ob, (long)N_*1024, 128, 1024,
                  N_, 128, N_, B_, H_);
    bgemm(stream, att, (long)H_*N_*N_, (long)N_*N_, N_,
                  kvpts + PQ_*3, (long)N_*480, 60, 480, 0, 1.0f,
                  opt, (long)N_*288, 36, 288,
                  N_, 36, N_, B_, H_);
    bgemm(stream, att, (long)H_*N_*N_, N_, N_*N_,
                  z, (long)N_*N_*128, (long)N_*128, 128, 0, 1.0f,
                  opair, (long)N_*1024, 1024, 128,
                  H_, 128, N_, B_, N_);
    inv_rot_k<<<eg(BN*96),256,0,stream>>>(opt, R, tr, opt2);
    ipa_cat_k<<<eg((long)BN*2432),256,0,stream>>>(ob, opt2, opair, cat);
    lin_launch(stream, cat, K.out, t1b, BN, 2432, 256, 0);
    scale_rows_k<<<eg((long)BN*256),256,0,stream>>>(t1b, mask, BN, 256);
    ln_k<<<dim3((unsigned)BN),32,0,stream>>>(s, t1b, K.ipa_ln.g, K.ipa_ln.b,
                                             nullptr, 0, s, 256);
    // -------- transformer --------
    hipMemcpyAsync(hbuf, s, (size_t)BN*256*sizeof(float),
                   hipMemcpyDeviceToDevice, stream);
    for (int l = 0; l < TFL; ++l) {
      TfLayer& T = K.tf[l];
      lin_launch(stream, hbuf, T.in_proj, qkv, BN, 256, 768, 0);
      bgemm(stream, qkv, (long)N_*768, 64, 768,
                    qkv+256, (long)N_*768, 64, 768, 1, 0.125f,
                    att4, (long)TFH*N_*N_, (long)N_*N_, N_,
                    N_, N_, 64, B_, TFH);
      key_pad_k<<<eg((long)B_*TFH*N_*N_),256,0,stream>>>(att4, mask);
      softmax_k<<<dim3((unsigned)(B_*TFH*N_)),32,0,stream>>>(att4, N_);
      bgemm(stream, att4, (long)TFH*N_*N_, (long)N_*N_, N_,
                    qkv+512, (long)N_*768, 64, 768, 0, 1.0f,
                    obuf4, (long)N_*256, 64, 256,
                    N_, 64, N_, B_, TFH);
      lin_launch(stream, obuf4, T.out_proj, t1b, BN, 256, 256, 0);
      ln_k<<<dim3((unsigned)BN),32,0,stream>>>(hbuf, t1b, T.ln1.g, T.ln1.b,
                                               nullptr, 0, hbuf, 256);
      lin_launch(stream, hbuf, T.ff1, t1b, BN, 256, 256, 1);
      lin_launch(stream, t1b,  T.ff2, t2b, BN, 256, 256, 0);
      ln_k<<<dim3((unsigned)BN),32,0,stream>>>(hbuf, t2b, T.ln2.g, T.ln2.b,
                                               nullptr, 0, hbuf, 256);
    }
    lin_launch(stream, hbuf, K.post, t1b, BN, 256, 256, 0);
    add_k<<<eg((long)BN*256),256,0,stream>>>(s, s, t1b, (long)BN*256);
    // -------- node transition --------
    lin_launch(stream, s,   K.tl1, t1b, BN, 256, 256, 1);
    lin_launch(stream, t1b, K.tl2, t2b, BN, 256, 256, 1);
    lin_launch(stream, t2b, K.tl3, t1b, BN, 256, 256, 0);
    ln_k<<<dim3((unsigned)BN),32,0,stream>>>(s, t1b, K.tln.g, K.tln.b,
                                             mask, 1, s, 256);
    // -------- backbone update --------
    lin_launch(stream, s, K.bb, updb, BN, 256, 6, 0);
    scale_rows_k<<<eg((long)BN*6),256,0,stream>>>(updb, mask, BN, 6);
    bb_update_k<<<eg(BN),256,0,stream>>>(updb, tr, R);
    // -------- edge transition --------
    if (K.has_et) {
      lin_launch(stream, s, K.et_init, neb, BN, 256, 128, 0);
      ecat_k<<<eg(BNN*384),256,0,stream>>>(z, neb, ecat);
      lin_launch(stream, ecat, K.et_t1, gh1, BNN, 384, 384, 1);
      lin_launch(stream, gh1,  K.et_t2, gh2, BNN, 384, 384, 1);
      add_k<<<eg(BNN*384),256,0,stream>>>(gh2, gh2, ecat, BNN*384);
      lin_launch(stream, gh2, K.et_final, eh1, BNN, 384, 128, 0);
      ln_k<<<dim3((unsigned)BNN),32,0,stream>>>(eh1, nullptr, K.et_ln.g, K.et_ln.b,
                                                mask, 2, z, 128);
    }
  }
  write_out_k<<<eg(BN*9),256,0,stream>>>(tr, R, (float*)d_out);
}
